// neural_nearest_neighbourblock_89919435309127
// MI455X (gfx1250) — compile-verified, matrix operand order checked by reading, not run on hardware
//
#include <hip/hip_runtime.h>
#include <cstdint>

// ---------------- problem constants ----------------
namespace cfg {
constexpr int B = 2, Cin = 8, H = 224, W = 224;
constexpr int P = 10, S = 5, K = 7, WIN = 3;
constexpr int E = 8, F = 64;
constexpr int N1 = (H - P) / S + 1;      // 43
constexpr int N2 = (W - P) / S + 1;      // 43
constexpr int N = N1 * N2;               // 1849
constexpr int M = (2 * WIN + 1) * (2 * WIN + 1); // 49
constexpr int D = Cin * P * P;           // 800
constexpr int HWsz = H * W;              // 50176
}

typedef __attribute__((ext_vector_type(16))) _Float16 v16h;
typedef __attribute__((ext_vector_type(8)))  float    v8f;

// ---------------- generic direct 3x3 SAME conv (f32, VALU) ----------------
__global__ __launch_bounds__(256)
void conv3x3_f32(const float* __restrict__ in, const float* __restrict__ wgt,
                 const float* __restrict__ bias, float* __restrict__ out,
                 int Cin_, int Cout_, int relu)
{
  int idx = blockIdx.x * blockDim.x + threadIdx.x;
  int total = cfg::B * Cout_ * cfg::HWsz;
  if (idx >= total) return;
  int w  = idx % cfg::W;
  int h  = (idx / cfg::W) % cfg::H;
  int co = (idx / cfg::HWsz) % Cout_;
  int b  = idx / (cfg::HWsz * Cout_);
  float acc = bias[co];
  const float* wco = wgt + (size_t)co * Cin_ * 9;
  for (int ci = 0; ci < Cin_; ++ci) {
    const float* xin = in + ((size_t)(b * Cin_ + ci)) * cfg::HWsz;
    const float* wk  = wco + ci * 9;
#pragma unroll
    for (int kh = 0; kh < 3; ++kh) {
      int r = h + kh - 1;
      if ((unsigned)r >= (unsigned)cfg::H) continue;
#pragma unroll
      for (int kw = 0; kw < 3; ++kw) {
        int c = w + kw - 1;
        if ((unsigned)c >= (unsigned)cfg::W) continue;
        acc += xin[r * cfg::W + c] * wk[kh * 3 + kw];
      }
    }
  }
  if (relu) acc = fmaxf(acc, 0.f);
  out[idx] = acc;
}

// ---------------- NCHW f32 -> NHWC f16 staging for WMMA A operand ----------------
__global__ __launch_bounds__(256)
void nchw_to_nhwc_f16(const float* __restrict__ in, _Float16* __restrict__ out)
{
  int idx = blockIdx.x * blockDim.x + threadIdx.x;
  int total = cfg::B * cfg::HWsz * cfg::F;
  if (idx >= total) return;
  int c   = idx % cfg::F;
  int pix = (idx / cfg::F) % cfg::HWsz;
  int b   = idx / (cfg::F * cfg::HWsz);
  out[idx] = (_Float16)in[((size_t)(b * cfg::F + c)) * cfg::HWsz + pix];
}

// ---------------- pre-swizzle 64x64x3x3 weights into per-lane WMMA B layout ----------------
// layout: [tap(9)][cinChunk(2)][coTile(4)][lane(32)][j(16)]
// lane L holds column N = L%16; element j maps to K = (j<8 ? 8*(L/16)+j : 16+8*(L/16)+j-8)
__global__ __launch_bounds__(256)
void pack_w2(const float* __restrict__ w, _Float16* __restrict__ out)
{
  int idx = blockIdx.x * blockDim.x + threadIdx.x;
  int total = 9 * 2 * 4 * 32 * 16;   // 36864
  if (idx >= total) return;
  int j    = idx & 15;
  int lane = (idx >> 4) & 31;
  int ct   = (idx >> 9) & 3;
  int cc   = (idx >> 11) & 1;
  int tap  = idx >> 12;
  int g = lane >> 4, n = lane & 15;
  int kloc = (j < 8) ? (8 * g + j) : (16 + 8 * g + (j - 8));
  int ci = cc * 32 + kloc;
  int co = ct * 16 + n;
  out[idx] = (_Float16)w[((size_t)co * cfg::F + ci) * 9 + tap];
}

// ---------------- conv2 (64->64, 3x3 SAME, relu) as implicit GEMM via WMMA ----------------
// one wave computes a 16-pixel x 16-outchannel tile; 9 taps x 2 cin-chunks = 18 wmma ops
__global__ __launch_bounds__(256)
void conv2_wmma(const _Float16* __restrict__ xh, const _Float16* __restrict__ wswz,
                const float* __restrict__ bias, float* __restrict__ out)
{
  constexpr int tilesX = cfg::W / 16;    // 14
  constexpr int coTiles = 4;
  int wave = (blockIdx.x * blockDim.x + threadIdx.x) >> 5;
  int lane = threadIdx.x & 31;
  int nTasks = cfg::B * cfg::H * tilesX * coTiles;
  if (wave >= nTasks) return;            // wave-uniform: EXEC stays all-ones for WMMA
  int ct = wave % coTiles;
  int pt = wave / coTiles;
  int w0 = (pt % tilesX) * 16;
  int h  = (pt / tilesX) % cfg::H;
  int b  = pt / (tilesX * cfg::H);
  int g = lane >> 4;
  int m = lane & 15;

  v8f acc = {0.f, 0.f, 0.f, 0.f, 0.f, 0.f, 0.f, 0.f};
#pragma unroll
  for (int tap = 0; tap < 9; ++tap) {
    int kh = tap / 3, kw = tap % 3;
    int r = h + kh - 1;
    if ((unsigned)r >= (unsigned)cfg::H) continue;
    int c = w0 + m + kw - 1;
    bool cOK = (unsigned)c < (unsigned)cfg::W;
    const _Float16* arow = xh + ((size_t)(b * cfg::H + r) * cfg::W + (cOK ? c : 0)) * cfg::F;
#pragma unroll
    for (int cc = 0; cc < 2; ++cc) {
      union { v16h v; uint4 u[2]; } a;
      if (cOK) {
        a.u[0] = *reinterpret_cast<const uint4*>(arow + cc * 32 + 8 * g);        // K j=0..7
        a.u[1] = *reinterpret_cast<const uint4*>(arow + cc * 32 + 16 + 8 * g);   // K j=8..15
      } else {
        a.u[0] = make_uint4(0u, 0u, 0u, 0u);
        a.u[1] = make_uint4(0u, 0u, 0u, 0u);
      }
      const v16h bmat = *reinterpret_cast<const v16h*>(
          wswz + ((size_t)(((tap * 2 + cc) * 4 + ct) * 32 + lane) << 4));
      acc = __builtin_amdgcn_wmma_f32_16x16x32_f16(
          /*neg_a=*/false, a.v, /*neg_b=*/false, bmat,
          /*c_mod=*/(short)0, acc, /*reuse_a=*/false, /*reuse_b=*/false);
    }
  }
  int co = ct * 16 + m;                    // D: lane%16 = column N = out channel
  float bs = bias[co];
  float* op = out + ((size_t)(b * cfg::F + co) * cfg::H + h) * cfg::W;
#pragma unroll
  for (int v = 0; v < 8; ++v) {            // D: row M = v + 8*(lane/16) = pixel
    int px = w0 + v + 8 * g;
    op[px] = fmaxf(acc[v] + bs, 0.f);
  }
}

// ---------------- distances + K-round softmax weights, one block per (b,n) ----------------
__global__ __launch_bounds__(128)
void dist_weights(const float* __restrict__ xe, const float* __restrict__ ltimg,
                  float* __restrict__ Wk)
{
  int bn = blockIdx.x;
  int b = bn / cfg::N, n = bn % cfg::N;
  int q1 = n / cfg::N2, q2 = n % cfg::N2;
  int r0 = q1 * cfg::S, c0 = q2 * cfg::S;
  int tid = threadIdx.x;
  int lane = tid & 31, wid = tid >> 5;

  __shared__ float sQ[cfg::D];
  __shared__ float sDot[cfg::M], sSq[cfg::M];
  __shared__ int   sCand[cfg::M];
  __shared__ float sRed[128];
  __shared__ float sCur[cfg::M], sWv[cfg::M];
  __shared__ float sScalar[2];

  for (int d = tid; d < cfg::D; d += 128) {
    int c = d / 100, rem = d % 100, pr = rem / 10, pc = rem % 10;
    sQ[d] = xe[((size_t)(b * cfg::E + c) * cfg::H + (r0 + pr)) * cfg::W + (c0 + pc)];
  }
  __syncthreads();

  if (wid == 0) {                          // ||pe_n||^2
    float s = 0.f;
    for (int d = lane; d < cfg::D; d += 32) s += sQ[d] * sQ[d];
    for (int o = 16; o > 0; o >>= 1) s += __shfl_xor(s, o, 32);
    if (lane == 0) sScalar[0] = s;
  }

  for (int mi = wid; mi < cfg::M; mi += 4) {
    int di = mi / 7 - cfg::WIN, dj = mi % 7 - cfg::WIN;
    int cq1 = min(max(q1 + di, 0), cfg::N1 - 1);
    int cq2 = min(max(q2 + dj, 0), cfg::N2 - 1);
    int cand = cq1 * cfg::N2 + cq2;
    int rr = cq1 * cfg::S, ccp = cq2 * cfg::S;
    float dot = 0.f, sq = 0.f;
    for (int d = lane; d < cfg::D; d += 32) {
      int c = d / 100, rem = d % 100, pr = rem / 10, pc = rem % 10;
      float y = xe[((size_t)(b * cfg::E + c) * cfg::H + (rr + pr)) * cfg::W + (ccp + pc)];
      dot += sQ[d] * y; sq += y * y;
    }
    for (int o = 16; o > 0; o >>= 1) {
      dot += __shfl_xor(dot, o, 32);
      sq  += __shfl_xor(sq,  o, 32);
    }
    if (lane == 0) { sDot[mi] = dot; sSq[mi] = sq; sCand[mi] = cand; }
  }

  if (tid < 100)                           // temperature patch mean
    sRed[tid] = ltimg[(size_t)b * cfg::HWsz + (r0 + tid / 10) * cfg::W + (c0 + tid % 10)];
  __syncthreads();
  if (tid == 0) {
    float s = 0.f;
    for (int i = 0; i < 100; ++i) s += sRed[i];
    sScalar[1] = expf(-s / 100.f);
  }
  __syncthreads();

  float scale = sScalar[1];
  float sqn = sScalar[0];
  if (tid < cfg::M) {
    float Dv = -(sqn + sSq[tid] - 2.f * sDot[tid]);
    sCur[tid] = (sCand[tid] == n) ? -1.0e9f : Dv * scale;
  }
  __syncthreads();

  float* wout = Wk + (size_t)(b * cfg::N + n) * cfg::M * cfg::K;
  for (int k = 0; k < cfg::K; ++k) {
    if (tid == 0) {
      float mx = -3.4e38f;
      for (int i = 0; i < cfg::M; ++i) mx = fmaxf(mx, sCur[i]);
      sScalar[0] = mx;
    }
    __syncthreads();
    if (tid < cfg::M) sWv[tid] = expf(sCur[tid] - sScalar[0]);
    __syncthreads();
    if (tid == 0) {
      float s = 0.f;
      for (int i = 0; i < cfg::M; ++i) s += sWv[i];
      sScalar[1] = s;
    }
    __syncthreads();
    if (tid < cfg::M) {
      float wv = sWv[tid] / sScalar[1];
      wout[tid * cfg::K + k] = wv;
      sCur[tid] += log1pf(-fminf(wv, 1.f - 1e-6f));
    }
    __syncthreads();
  }
}

// ---------------- weighted aggregation + overlap-add scatter, one block per (b,n) ----------------
__global__ __launch_bounds__(128)
void aggregate(const float* __restrict__ x, const float* __restrict__ Wk,
               float* __restrict__ accum)
{
  int bn = blockIdx.x;
  int b = bn / cfg::N, n = bn % cfg::N;
  int q1 = n / cfg::N2, q2 = n % cfg::N2;
  int r0 = q1 * cfg::S, c0 = q2 * cfg::S;
  int tid = threadIdx.x;
  __shared__ float sW[cfg::M * cfg::K];
  __shared__ int sR[cfg::M], sC[cfg::M];
  const float* wsrc = Wk + (size_t)(b * cfg::N + n) * cfg::M * cfg::K;
  for (int t = tid; t < cfg::M * cfg::K; t += 128) sW[t] = wsrc[t];
  if (tid < cfg::M) {
    int di = tid / 7 - cfg::WIN, dj = tid % 7 - cfg::WIN;
    int cq1 = min(max(q1 + di, 0), cfg::N1 - 1);
    int cq2 = min(max(q2 + dj, 0), cfg::N2 - 1);
    sR[tid] = cq1 * cfg::S; sC[tid] = cq2 * cfg::S;
  }
  __syncthreads();
  for (int d = tid; d < cfg::D; d += 128) {
    int c = d / 100, rem = d % 100, pr = rem / 10, pc = rem % 10;
    float acc[cfg::K];
#pragma unroll
    for (int k = 0; k < cfg::K; ++k) acc[k] = 0.f;
    const float* xb = x + (size_t)(b * cfg::Cin + c) * cfg::HWsz;
    for (int mi = 0; mi < cfg::M; ++mi) {
      float y = xb[(sR[mi] + pr) * cfg::W + (sC[mi] + pc)];
#pragma unroll
      for (int k = 0; k < cfg::K; ++k) acc[k] += sW[mi * cfg::K + k] * y;
    }
    int opix = (r0 + pr) * cfg::W + (c0 + pc);
#pragma unroll
    for (int k = 0; k < cfg::K; ++k)
      atomicAdd(&accum[((size_t)(k * cfg::B + b) * cfg::Cin + c) * cfg::HWsz + opix], acc[k]);
  }
}

// ---------------- final: concat x with count-normalized folded volumes ----------------
__device__ __forceinline__ int coverage1d(int h)
{
  int hi = h / cfg::S; if (hi > cfg::N1 - 1) hi = cfg::N1 - 1;
  int lo = (h > cfg::P - 1) ? ((h - (cfg::P - 1) + cfg::S - 1) / cfg::S) : 0;
  return (hi >= lo) ? (hi - lo + 1) : 0;
}

__global__ __launch_bounds__(256)
void finalize(const float* __restrict__ x, const float* __restrict__ accum,
              float* __restrict__ out)
{
  constexpr int OC = cfg::Cin * (cfg::K + 1);   // 64
  int idx = blockIdx.x * blockDim.x + threadIdx.x;
  int total = cfg::B * OC * cfg::HWsz;
  if (idx >= total) return;
  int wpx = idx % cfg::W;
  int hpx = (idx / cfg::W) % cfg::H;
  int ch  = (idx / cfg::HWsz) % OC;
  int b   = idx / (cfg::HWsz * OC);
  float v;
  if (ch < cfg::Cin) {
    v = x[(size_t)(b * cfg::Cin + ch) * cfg::HWsz + hpx * cfg::W + wpx];
  } else {
    int k = (ch - cfg::Cin) / cfg::Cin;
    int c = (ch - cfg::Cin) % cfg::Cin;
    float cnt = (float)(coverage1d(hpx) * coverage1d(wpx));
    float a = accum[((size_t)(k * cfg::B + b) * cfg::Cin + c) * cfg::HWsz + hpx * cfg::W + wpx];
    v = a / fmaxf(cnt, 1.f);
  }
  out[idx] = v;
}

// ---------------- host launcher ----------------
extern "C" void kernel_launch(void* const* d_in, const int* in_sizes, int n_in,
                              void* d_out, int out_size, void* d_ws, size_t ws_size,
                              hipStream_t stream)
{
  (void)in_sizes; (void)n_in; (void)out_size; (void)ws_size;
  using namespace cfg;
  const float* x   = (const float*)d_in[0];
  const float* w1e = (const float*)d_in[1];
  const float* b1e = (const float*)d_in[2];
  const float* w2e = (const float*)d_in[3];
  const float* b2e = (const float*)d_in[4];
  const float* w3e = (const float*)d_in[5];
  const float* b3e = (const float*)d_in[6];
  const float* w1t = (const float*)d_in[7];
  const float* b1t = (const float*)d_in[8];
  const float* w2t = (const float*)d_in[9];
  const float* b2t = (const float*)d_in[10];
  const float* w3t = (const float*)d_in[11];
  const float* b3t = (const float*)d_in[12];
  float* out = (float*)d_out;

  char* ws = (char*)d_ws;
  size_t off = 0;
  auto alloc = [&](size_t bytes) -> char* {
    char* p = ws + off;
    off += (bytes + 255) & ~(size_t)255;
    return p;
  };
  float*    h1    = (float*)alloc(sizeof(float) * (size_t)B * F * HWsz);
  float*    h2    = (float*)alloc(sizeof(float) * (size_t)B * F * HWsz);
  float*    xe    = (float*)alloc(sizeof(float) * (size_t)B * E * HWsz);
  float*    ltimg = (float*)alloc(sizeof(float) * (size_t)B * HWsz);
  float*    Wkb   = (float*)alloc(sizeof(float) * (size_t)B * N * M * K);
  float*    accum = (float*)alloc(sizeof(float) * (size_t)K * B * Cin * HWsz);
  _Float16* h1h   = (_Float16*)alloc(sizeof(_Float16) * (size_t)B * F * HWsz);
  _Float16* wswz  = (_Float16*)alloc(sizeof(_Float16) * 36864);

  hipMemsetAsync(accum, 0, sizeof(float) * (size_t)K * B * Cin * HWsz, stream);

  auto blocks = [](long long total, int tpb) { return (int)((total + tpb - 1) / tpb); };
  const long long nFull = (long long)B * F * HWsz;
  const int wmmaWaves = B * H * (W / 16) * 4;

  // ---- embedding net: 8 -> 64 -> 64 -> 8 ----
  conv3x3_f32<<<blocks((long long)B * F * HWsz, 256), 256, 0, stream>>>(x, w1e, b1e, h1, Cin, F, 1);
  nchw_to_nhwc_f16<<<blocks(nFull, 256), 256, 0, stream>>>(h1, h1h);
  pack_w2<<<blocks(36864, 256), 256, 0, stream>>>(w2e, wswz);
  conv2_wmma<<<blocks((long long)wmmaWaves * 32, 256), 256, 0, stream>>>(h1h, wswz, b2e, h2);
  conv3x3_f32<<<blocks((long long)B * E * HWsz, 256), 256, 0, stream>>>(h2, w3e, b3e, xe, F, E, 0);

  // ---- temperature net: 8 -> 64 -> 64 -> 1 ----
  conv3x3_f32<<<blocks((long long)B * F * HWsz, 256), 256, 0, stream>>>(x, w1t, b1t, h1, Cin, F, 1);
  nchw_to_nhwc_f16<<<blocks(nFull, 256), 256, 0, stream>>>(h1, h1h);
  pack_w2<<<blocks(36864, 256), 256, 0, stream>>>(w2t, wswz);
  conv2_wmma<<<blocks((long long)wmmaWaves * 32, 256), 256, 0, stream>>>(h1h, wswz, b2t, h2);
  conv3x3_f32<<<blocks((long long)B * 1 * HWsz, 256), 256, 0, stream>>>(h2, w3t, b3t, ltimg, F, 1, 0);

  // ---- neighbourhood weights, aggregation, fold, concat ----
  dist_weights<<<B * N, 128, 0, stream>>>(xe, ltimg, Wkb);
  aggregate<<<B * N, 128, 0, stream>>>(x, Wkb, accum);
  finalize<<<blocks((long long)B * Cin * (K + 1) * HWsz, 256), 256, 0, stream>>>(x, accum, out);
}
